// SpatioTemporalGNN_13597866459874
// MI455X (gfx1250) — compile-verified, matrix operand order checked by reading, not run on hardware
//
#include <hip/hip_runtime.h>
#include <hip/hip_bf16.h>

// ---------------------------------------------------------------------------
// SpatioTemporalGNN for MI455X (gfx1250).
//   - GRU gate GEMMs run on v_wmma_f32_16x16x32_f16 (f16 in, f32 accum).
//   - Recurrent h kept in WMMA C-layout registers; per-wave LDS tile does the
//     C-layout -> A-layout cross-lane transpose between timesteps.
//   - SAGE mean aggregation via global f32 atomics; layernorm via LDS reduce.
// ---------------------------------------------------------------------------

#define N_NODES 50000
#define SEQ_T   16
#define N_FEATK 8
#define HID     64
#define N_EDGES 1600000

typedef __attribute__((ext_vector_type(16))) _Float16 v16h;
typedef __attribute__((ext_vector_type(8)))  _Float16 v8h;
typedef __attribute__((ext_vector_type(8)))  float    v8f;

union Frag16 { v16h v; v8h h[2]; };

__device__ __forceinline__ v16h load_frag2(const _Float16* p0, const _Float16* p1) {
  Frag16 f;
  f.h[0] = *(const v8h*)p0;   // 16B -> *_load_b128
  f.h[1] = *(const v8h*)p1;
  return f.v;
}

// A fragment (16x32 f16, M x K). lane l<16: M=l, halves K=k0+{0..7},{16..23};
// lane l+16: same M, halves K=k0+{8..15},{24..31}.  rowPtr points at row M's k=0.
__device__ __forceinline__ v16h load_fragA(const _Float16* rowPtr, int koff, int hi) {
  const _Float16* p = rowPtr + koff + hi * 8;
  return load_frag2(p, p + 16);
}

// B fragment (32x16 f16, K x N) for D = A*B with B[k][n] = W[n][k] (row-major W,
// row pitch `pitch` halves). lane l<16: N=l; contiguous k-runs from W row n.
__device__ __forceinline__ v16h load_fragB(const _Float16* W, int nrow, int pitch,
                                           int koff, int hi) {
  const _Float16* p = W + (size_t)nrow * pitch + koff + hi * 8;
  return load_frag2(p, p + 16);
}

__device__ __forceinline__ v8f wmma_f16f32(v16h a, v16h b, v8f c) {
  // (neg_a, A, neg_b, B, c_mod, C, reuse_a, reuse_b)
  return __builtin_amdgcn_wmma_f32_16x16x32_f16(false, a, false, b, (short)0, c,
                                                false, false);
}

__device__ __forceinline__ float sigmoidf_(float x) {
  return 1.0f / (1.0f + __expf(-x));
}

// ---------------------------------------------------------------------------
// Fused GRU direction kernel.
//   xin  : (nn, 16, IND) f16       Wih16: (192, IND) f16   Whh16: (192, 64) f16
//   WRITE_ALL:  write each step's h (f16) into outAll[(node*16+t)*128 + outoff + j]
//   !WRITE_ALL: write only the last processed step's h (f32) into
//               outLast[node*128 + outoff + j]
// One wave handles 16 nodes; 4 waves per workgroup.
// ---------------------------------------------------------------------------
template <int IND, bool REV, bool WRITE_ALL>
__global__ __launch_bounds__(128, 1) void gru_dir_kernel(
    const _Float16* __restrict__ xin,
    const _Float16* __restrict__ Wih16,
    const _Float16* __restrict__ Whh16,
    const float* __restrict__ bih,
    const float* __restrict__ bhh,
    _Float16* __restrict__ outAll,
    float* __restrict__ outLast,
    int outoff, int nsteps, int nn)
{
  constexpr int NXC = IND / 32;                 // k-chunks over the input part
  __shared__ _Float16 sH[4][16 * 64];           // per-wave h staging (transpose)

  const int tid  = threadIdx.x;
  const int wave = tid >> 5;
  const int lane = tid & 31;
  const int np   = lane & 15;                   // M (A-frag) / N (B-frag) index
  const int hi   = lane >> 4;                   // half-select within K
  const int nodeBase = (blockIdx.x * 4 + wave) * 16;
  int rowc = nodeBase + np;
  if (rowc > nn - 1) rowc = nn - 1;             // clamp (keeps EXEC uniform)
  _Float16* hS = &sH[wave][0];

  // Per-lane gate biases (j = jt*16 + np), hoisted out of the time loop.
  float br[4], bz[4], bni[4], bnh[4];
#pragma unroll
  for (int jt = 0; jt < 4; jt++) {
    int j   = jt * 16 + np;
    br[jt]  = bih[j] + bhh[j];
    bz[jt]  = bih[64 + j] + bhh[64 + j];
    bni[jt] = bih[128 + j];
    bnh[jt] = bhh[128 + j];
  }

  // h state in WMMA C layout: hreg[jt][r] = h[m = r+8*hi][j = jt*16+np]
  float hreg[4][8];
#pragma unroll
  for (int jt = 0; jt < 4; jt++)
#pragma unroll
    for (int r = 0; r < 8; r++) hreg[jt][r] = 0.0f;
#pragma unroll
  for (int jt = 0; jt < 4; jt++)
#pragma unroll
    for (int r = 0; r < 8; r++)
      hS[(r + 8 * hi) * 64 + jt * 16 + np] = (_Float16)0.0f;

  for (int i = 0; i < nsteps; i++) {
    const int t = REV ? (SEQ_T - 1 - i) : i;

    // A fragments: input row (global) and recurrent h (LDS staging)
    const _Float16* xrow = xin + ((size_t)rowc * SEQ_T + t) * IND;
    v16h ax[NXC];
#pragma unroll
    for (int c = 0; c < NXC; c++) ax[c] = load_fragA(xrow, 32 * c, hi);
    const _Float16* hrow = hS + np * 64;
    v16h ah[2];
#pragma unroll
    for (int c = 0; c < 2; c++) ah[c] = load_fragA(hrow, 32 * c, hi);

    // r/z gates: fused accumulation of x*Wih + h*Whh over N = 0..127
    v8f accrz[8];
#pragma unroll
    for (int nt = 0; nt < 8; nt++) {
      v8f acc = {0.f, 0.f, 0.f, 0.f, 0.f, 0.f, 0.f, 0.f};
#pragma unroll
      for (int c = 0; c < NXC; c++)
        acc = wmma_f16f32(ax[c], load_fragB(Wih16, nt * 16 + np, IND, 32 * c, hi), acc);
#pragma unroll
      for (int c = 0; c < 2; c++)
        acc = wmma_f16f32(ah[c], load_fragB(Whh16, nt * 16 + np, 64, 32 * c, hi), acc);
      accrz[nt] = acc;
    }

    // n gate needs inn (input part) and hn (recurrent part) SEPARATELY.
    v8f accin[4], acchn[4];
#pragma unroll
    for (int nt = 0; nt < 4; nt++) {
      v8f ai = {0.f, 0.f, 0.f, 0.f, 0.f, 0.f, 0.f, 0.f};
      v8f ar = {0.f, 0.f, 0.f, 0.f, 0.f, 0.f, 0.f, 0.f};
#pragma unroll
      for (int c = 0; c < NXC; c++)
        ai = wmma_f16f32(ax[c], load_fragB(Wih16, 128 + nt * 16 + np, IND, 32 * c, hi), ai);
#pragma unroll
      for (int c = 0; c < 2; c++)
        ar = wmma_f16f32(ah[c], load_fragB(Whh16, 128 + nt * 16 + np, 64, 32 * c, hi), ar);
      accin[nt] = ai;
      acchn[nt] = ar;
    }

    // Gate nonlinearity + state update (lane-local: C-layout positions align).
#pragma unroll
    for (int jt = 0; jt < 4; jt++)
#pragma unroll
      for (int r = 0; r < 8; r++) {
        float rg = sigmoidf_(accrz[jt][r] + br[jt]);
        float zg = sigmoidf_(accrz[4 + jt][r] + bz[jt]);
        float ng = tanhf(accin[jt][r] + bni[jt] + rg * (acchn[jt][r] + bnh[jt]));
        hreg[jt][r] = (1.0f - zg) * ng + zg * hreg[jt][r];
      }

    // Stage new h for next step's A-fragments; emit per-step output if needed.
#pragma unroll
    for (int jt = 0; jt < 4; jt++)
#pragma unroll
      for (int r = 0; r < 8; r++) {
        const int m = r + 8 * hi;
        const int j = jt * 16 + np;
        _Float16 hh = (_Float16)hreg[jt][r];
        hS[m * 64 + j] = hh;
        if (WRITE_ALL) {
          const int node = nodeBase + m;
          if (node < nn)
            outAll[((size_t)node * SEQ_T + t) * 128 + outoff + j] = hh;
        }
      }

    if (!WRITE_ALL && i == nsteps - 1) {
#pragma unroll
      for (int jt = 0; jt < 4; jt++)
#pragma unroll
        for (int r = 0; r < 8; r++) {
          const int node = nodeBase + r + 8 * hi;
          if (node < nn)
            outLast[(size_t)node * 128 + outoff + jt * 16 + np] = hreg[jt][r];
        }
    }
  }
}

// ---------------------------------------------------------------------------
// Supporting kernels (VALU / atomics).
// ---------------------------------------------------------------------------
__global__ void f32_to_f16_kernel(const float* __restrict__ src,
                                  _Float16* __restrict__ dst, int n) {
  int i = blockIdx.x * blockDim.x + threadIdx.x;
  if (i < n) dst[i] = (_Float16)src[i];
}

__global__ void zero_kernel(float* __restrict__ p, size_t n) {
  size_t i = (size_t)blockIdx.x * blockDim.x + threadIdx.x;
  size_t stride = (size_t)gridDim.x * blockDim.x;
  for (; i < n; i += stride) p[i] = 0.0f;
}

// h0[n,t,o] = x[n,t,:] @ Wip.T + bip  (f16 output for the GRU)
__global__ void input_proj_kernel(const float* __restrict__ x,
                                  const float* __restrict__ W,
                                  const float* __restrict__ b,
                                  _Float16* __restrict__ h0, int rows) {
  int idx = blockIdx.x * blockDim.x + threadIdx.x;
  if (idx >= rows * 64) return;
  int o = idx & 63, r = idx >> 6;
  const float* xr = x + (size_t)r * N_FEATK;
  float acc = b[o];
#pragma unroll
  for (int f = 0; f < N_FEATK; f++) acc += xr[f] * W[o * N_FEATK + f];
  h0[idx] = (_Float16)acc;
}

// hnode[n,o] = hlast[n,:] @ Wop.T + bop  (only t=15 of layer-2 output exists)
__global__ void output_proj_kernel(const float* __restrict__ hl,
                                   const float* __restrict__ W,
                                   const float* __restrict__ b,
                                   float* __restrict__ hn, int N) {
  int idx = blockIdx.x * blockDim.x + threadIdx.x;
  if (idx >= N * 64) return;
  int o = idx & 63, n = idx >> 6;
  const float* hr = hl + (size_t)n * 128;
  float acc = b[o];
  for (int j = 0; j < 128; j++) acc += hr[j] * W[o * 128 + j];
  hn[idx] = acc;
}

__global__ void sage_scatter_kernel(const int* __restrict__ src,
                                    const int* __restrict__ dst,
                                    const float* __restrict__ h,
                                    float* __restrict__ agg,
                                    float* __restrict__ cnt, int E) {
  int e = blockIdx.x * blockDim.x + threadIdx.x;
  if (e >= E) return;
  int s = src[e], d = dst[e];
  atomicAdd(&cnt[d], 1.0f);
  const float* hr = h + (size_t)s * 64;
  float* ar = agg + (size_t)d * 64;
  for (int j = 0; j < 64; j++) atomicAdd(&ar[j], hr[j]);
}

// h[n] = h[n] + relu(layernorm(agg/cnt @ Wl.T + bl + h @ Wr.T))
__global__ __launch_bounds__(64) void sage_update_kernel(
    const float* __restrict__ agg, const float* __restrict__ cnt,
    const float* __restrict__ Wl, const float* __restrict__ bl,
    const float* __restrict__ Wr, const float* __restrict__ g,
    const float* __restrict__ be, float* __restrict__ h, int N) {
  __shared__ float hs[64], as[64], red[64];
  int j = threadIdx.x;
  int n = blockIdx.x;
  hs[j] = h[(size_t)n * 64 + j];
  float c = cnt[n];
  if (c < 1.0f) c = 1.0f;
  as[j] = agg[(size_t)n * 64 + j] / c;
  __syncthreads();
  float acc = bl[j];
  for (int k = 0; k < 64; k++) acc += as[k] * Wl[j * 64 + k] + hs[k] * Wr[j * 64 + k];
  red[j] = acc;
  __syncthreads();
  for (int s = 32; s > 0; s >>= 1) { if (j < s) red[j] += red[j + s]; __syncthreads(); }
  float mean = red[0] * (1.0f / 64.0f);
  __syncthreads();
  float d0 = acc - mean;
  red[j] = d0 * d0;
  __syncthreads();
  for (int s = 32; s > 0; s >>= 1) { if (j < s) red[j] += red[j + s]; __syncthreads(); }
  float var = red[0] * (1.0f / 64.0f);
  float y = d0 * rsqrtf(var + 1e-5f) * g[j] + be[j];
  y = y > 0.0f ? y : 0.0f;
  h[(size_t)n * 64 + j] = hs[j] + y;
}

__global__ __launch_bounds__(64) void pool_partial_kernel(
    const float* __restrict__ h, float* __restrict__ ghsum, int N) {
  int j = threadIdx.x;
  int nb = gridDim.x;
  int chunk = (N + nb - 1) / nb;
  int n0 = blockIdx.x * chunk;
  int n1 = n0 + chunk; if (n1 > N) n1 = N;
  float acc = 0.0f;
  for (int n = n0; n < n1; n++) acc += h[(size_t)n * 64 + j];
  atomicAdd(&ghsum[j], acc);
}

// gh MLP + impact head (single block)
__global__ __launch_bounds__(64) void heads_global_kernel(
    const float* __restrict__ ghsum,
    const float* __restrict__ W1, const float* __restrict__ b1,
    const float* __restrict__ W2, const float* __restrict__ b2,
    const float* __restrict__ Wi1, const float* __restrict__ bi1,
    const float* __restrict__ Wi2, const float* __restrict__ bi2,
    float* __restrict__ gh2, float* __restrict__ out_imp,
    float* __restrict__ out_gh, int N) {
  __shared__ float s0[64], s1[32], s2[64], s3[32];
  int j = threadIdx.x;
  s0[j] = ghsum[j] / (float)N;
  __syncthreads();
  if (j < 32) {
    float a = b1[j];
    for (int k = 0; k < 64; k++) a += W1[j * 64 + k] * s0[k];
    s1[j] = a > 0.0f ? a : 0.0f;
  }
  __syncthreads();
  {
    float a = b2[j];
    for (int k = 0; k < 32; k++) a += W2[j * 32 + k] * s1[k];
    s2[j] = a; gh2[j] = a; out_gh[j] = a;
  }
  __syncthreads();
  if (j < 32) {
    float a = bi1[j];
    for (int k = 0; k < 64; k++) a += Wi1[j * 64 + k] * s2[k];
    s3[j] = a > 0.0f ? a : 0.0f;
  }
  __syncthreads();
  if (j < 3) {
    float a = bi2[j];
    for (int k = 0; k < 32; k++) a += Wi2[j * 32 + k] * s3[k];
    out_imp[j] = a;
  }
}

// severity + spatial heads, one thread per node; also copies h into d_out
__global__ __launch_bounds__(128) void heads_node_kernel(
    const float* __restrict__ h, const float* __restrict__ gh2,
    const float* __restrict__ Ws1, const float* __restrict__ bs1,
    const float* __restrict__ Ws2, const float* __restrict__ bs2,
    const float* __restrict__ We1, const float* __restrict__ be1,
    const float* __restrict__ We2, const float* __restrict__ be2,
    float* __restrict__ out_sev, float* __restrict__ out_ext,
    float* __restrict__ out_h, int N) {
  int n = blockIdx.x * blockDim.x + threadIdx.x;
  if (n >= N) return;
  float hr[64];
#pragma unroll
  for (int j = 0; j < 64; j++) {
    hr[j] = h[(size_t)n * 64 + j];
    out_h[(size_t)n * 64 + j] = hr[j];
  }
  float s = bs2[0];
  for (int k = 0; k < 32; k++) {
    float a = bs1[k];
#pragma unroll
    for (int j = 0; j < 64; j++) a += Ws1[k * 64 + j] * hr[j];
    a = a > 0.0f ? a : 0.0f;
    s += Ws2[k] * a;
  }
  out_sev[n] = s > 0.0f ? s : 0.0f;
  float e = be2[0];
  for (int k = 0; k < 32; k++) {
    float a = be1[k];
#pragma unroll
    for (int j = 0; j < 64; j++) a += We1[k * 128 + j] * hr[j];
#pragma unroll
    for (int j = 0; j < 64; j++) a += We1[k * 128 + 64 + j] * gh2[j];
    a = a > 0.0f ? a : 0.0f;
    e += We2[k] * a;
  }
  out_ext[n] = 1.0f / (1.0f + __expf(-e));
}

// ---------------------------------------------------------------------------
extern "C" void kernel_launch(void* const* d_in, const int* in_sizes, int n_in,
                              void* d_out, int out_size, void* d_ws, size_t ws_size,
                              hipStream_t stream) {
  (void)in_sizes; (void)n_in; (void)out_size; (void)ws_size;
  const int N = N_NODES, T = SEQ_T, E = N_EDGES;

  // ---- inputs (setup_inputs insertion order, params flattened recursively) --
  const float* x    = (const float*)d_in[0];
  const int*   ei   = (const int*)d_in[1];
  const int*   esrc = ei;
  const int*   edst = ei + E;
  const float* Wip = (const float*)d_in[2];
  const float* bip = (const float*)d_in[3];
  const float* Wih0f = (const float*)d_in[4];  const float* Whh0f = (const float*)d_in[5];
  const float* bih0f = (const float*)d_in[6];  const float* bhh0f = (const float*)d_in[7];
  const float* Wih0b = (const float*)d_in[8];  const float* Whh0b = (const float*)d_in[9];
  const float* bih0b = (const float*)d_in[10]; const float* bhh0b = (const float*)d_in[11];
  const float* Wih1f = (const float*)d_in[12]; const float* Whh1f = (const float*)d_in[13];
  const float* bih1f = (const float*)d_in[14]; const float* bhh1f = (const float*)d_in[15];
  const float* Wih1b = (const float*)d_in[16]; const float* Whh1b = (const float*)d_in[17];
  const float* bih1b = (const float*)d_in[18]; const float* bhh1b = (const float*)d_in[19];
  const float* Wop = (const float*)d_in[20];
  const float* bop = (const float*)d_in[21];
  // gnn layers at d_in[22 + 5*l ..], global_pool at 37.., severity 41..,
  // impact 45.., spatial 49..
  const float* Wg1 = (const float*)d_in[37]; const float* bg1 = (const float*)d_in[38];
  const float* Wg2 = (const float*)d_in[39]; const float* bg2 = (const float*)d_in[40];
  const float* Ws1 = (const float*)d_in[41]; const float* bs1 = (const float*)d_in[42];
  const float* Ws2 = (const float*)d_in[43]; const float* bs2 = (const float*)d_in[44];
  const float* Wi1 = (const float*)d_in[45]; const float* bi1 = (const float*)d_in[46];
  const float* Wi2 = (const float*)d_in[47]; const float* bi2 = (const float*)d_in[48];
  const float* We1 = (const float*)d_in[49]; const float* be1 = (const float*)d_in[50];
  const float* We2 = (const float*)d_in[51]; const float* be2 = (const float*)d_in[52];

  // ---- workspace carve (~360 MB total) -------------------------------------
  char* ws = (char*)d_ws;
  size_t off = 0;
  auto carve = [&](size_t bytes) -> char* {
    char* p = ws + off;
    off += (bytes + 255) & ~(size_t)255;
    return p;
  };
  _Float16* h0    = (_Float16*)carve((size_t)N * T * 64 * 2);   // 102.4 MB
  _Float16* bufA  = (_Float16*)carve((size_t)N * T * 128 * 2);  // 204.8 MB
  float*    hlast = (float*)carve((size_t)N * 128 * 4);         // 25.6 MB
  float*    hnode = (float*)carve((size_t)N * 64 * 4);          // 12.8 MB
  float*    agg   = (float*)carve((size_t)N * 65 * 4);          // agg + cnt
  float*    cnt   = agg + (size_t)N * 64;
  float*    ghsum = (float*)carve(64 * 4);
  float*    gh2   = (float*)carve(64 * 4);
  _Float16* fWih0f = (_Float16*)carve(192 * 64 * 2);
  _Float16* fWhh0f = (_Float16*)carve(192 * 64 * 2);
  _Float16* fWih0b = (_Float16*)carve(192 * 64 * 2);
  _Float16* fWhh0b = (_Float16*)carve(192 * 64 * 2);
  _Float16* fWih1f = (_Float16*)carve(192 * 128 * 2);
  _Float16* fWhh1f = (_Float16*)carve(192 * 64 * 2);
  _Float16* fWih1b = (_Float16*)carve(192 * 128 * 2);
  _Float16* fWhh1b = (_Float16*)carve(192 * 64 * 2);

  // ---- weights -> f16 ------------------------------------------------------
  auto conv = [&](const float* s, _Float16* d, int n) {
    f32_to_f16_kernel<<<(n + 255) / 256, 256, 0, stream>>>(s, d, n);
  };
  conv(Wih0f, fWih0f, 192 * 64);  conv(Whh0f, fWhh0f, 192 * 64);
  conv(Wih0b, fWih0b, 192 * 64);  conv(Whh0b, fWhh0b, 192 * 64);
  conv(Wih1f, fWih1f, 192 * 128); conv(Whh1f, fWhh1f, 192 * 64);
  conv(Wih1b, fWih1b, 192 * 128); conv(Whh1b, fWhh1b, 192 * 64);

  // ---- input projection ----------------------------------------------------
  {
    int total = N * T * 64;
    input_proj_kernel<<<(total + 255) / 256, 256, 0, stream>>>(x, Wip, bip, h0, N * T);
  }

  // ---- bidirectional 2-layer GRU (WMMA) ------------------------------------
  const int gblocks = (N + 63) / 64;
  gru_dir_kernel<64, false, true><<<gblocks, 128, 0, stream>>>(
      h0, fWih0f, fWhh0f, bih0f, bhh0f, bufA, nullptr, 0, 16, N);
  gru_dir_kernel<64, true, true><<<gblocks, 128, 0, stream>>>(
      h0, fWih0b, fWhh0b, bih0b, bhh0b, bufA, nullptr, 64, 16, N);
  gru_dir_kernel<128, false, false><<<gblocks, 128, 0, stream>>>(
      bufA, fWih1f, fWhh1f, bih1f, bhh1f, nullptr, hlast, 0, 16, N);
  // reverse direction at t=15 is its FIRST scan step -> one step suffices
  gru_dir_kernel<128, true, false><<<gblocks, 128, 0, stream>>>(
      bufA, fWih1b, fWhh1b, bih1b, bhh1b, nullptr, hlast, 64, 1, N);

  // ---- output projection (last timestep only) ------------------------------
  output_proj_kernel<<<(N * 64 + 255) / 256, 256, 0, stream>>>(hlast, Wop, bop, hnode, N);

  // ---- 3x SAGE + layernorm + relu + residual -------------------------------
  for (int l = 0; l < 3; l++) {
    const float* Wl = (const float*)d_in[22 + 5 * l];
    const float* bl = (const float*)d_in[23 + 5 * l];
    const float* Wr = (const float*)d_in[24 + 5 * l];
    const float* g  = (const float*)d_in[25 + 5 * l];
    const float* be = (const float*)d_in[26 + 5 * l];
    zero_kernel<<<2048, 256, 0, stream>>>(agg, (size_t)N * 65);
    sage_scatter_kernel<<<(E + 255) / 256, 256, 0, stream>>>(esrc, edst, hnode, agg, cnt, E);
    sage_update_kernel<<<N, 64, 0, stream>>>(agg, cnt, Wl, bl, Wr, g, be, hnode, N);
  }

  // ---- global pool + heads -------------------------------------------------
  float* out     = (float*)d_out;
  float* out_sev = out;
  float* out_imp = out + N;
  float* out_ext = out + N + 3;
  float* out_h   = out + 2 * (size_t)N + 3;
  float* out_gh  = out + 2 * (size_t)N + 3 + (size_t)N * 64;

  zero_kernel<<<1, 64, 0, stream>>>(ghsum, 64);
  pool_partial_kernel<<<200, 64, 0, stream>>>(hnode, ghsum, N);
  heads_global_kernel<<<1, 64, 0, stream>>>(ghsum, Wg1, bg1, Wg2, bg2,
                                            Wi1, bi1, Wi2, bi2,
                                            gh2, out_imp, out_gh, N);
  heads_node_kernel<<<(N + 127) / 128, 128, 0, stream>>>(
      hnode, gh2, Ws1, bs1, Ws2, bs2, We1, be1, We2, be2,
      out_sev, out_ext, out_h, N);
}
